// BasicConceptQuantizationV4Smooth_8203387535673
// MI455X (gfx1250) — matrix operand order repaired and checked by reading, not compiled, not run on hardware
//
#include <hip/hip_runtime.h>
#include <cmath>

// ---------------------------------------------------------------------------
// Problem sizes (fixed by the reference)
// ---------------------------------------------------------------------------
constexpr int Bn = 16384;   // batch
constexpr int Dd = 768;     // feature dim
constexpr int Kk = 512;     // concepts
constexpr int Cc = 1000;    // classes
constexpr int Cp = 1008;    // classes padded to multiple of 16
constexpr float SMOOTHING   = 0.1f;
constexpr float GRAD_FACTOR = 0.1f;
constexpr float LN_EPS      = 1e-5f;

// ---------------------------------------------------------------------------
// Vector types for CDNA5 WMMA
// ---------------------------------------------------------------------------
typedef __attribute__((ext_vector_type(16))) __bf16 v16bf;
typedef __attribute__((ext_vector_type(8)))  __bf16 v8bf;
typedef __attribute__((ext_vector_type(8)))  float  v8f;

union Frag16 { v16bf v; v8bf h[2]; };

// ---------------------------------------------------------------------------
// Fragment loaders (CDNA5 ISA 7.12.2 layouts)
//   A 16x32 bf16: lanes 0-15 -> M=lane, K {0..7,16..23}; lanes 16-31 -> {8..15,24..31}
//   B 32x16 bf16: lanes 0-15 -> N=lane, K {0..7,16..23}; lanes 16-31 -> {8..15,24..31}
// aBase/bBase already include the per-lane row and half offsets.
// ---------------------------------------------------------------------------
template <int KD>
__device__ __forceinline__ void load_frags(const __bf16* __restrict__ aBase,
                                           const __bf16* __restrict__ bBase,
                                           int k0, Frag16& bf, Frag16 (&af)[4])
{
    bf.h[0] = *(const v8bf*)(bBase + k0);
    bf.h[1] = *(const v8bf*)(bBase + k0 + 16);
#pragma unroll
    for (int i = 0; i < 4; ++i) {
        const __bf16* ap = aBase + (size_t)i * (16 * KD) + k0;
        af[i].h[0] = *(const v8bf*)(ap);        // K = k0 + half*8 + 0..7
        af[i].h[1] = *(const v8bf*)(ap + 16);   // K = k0 + 16 + half*8 + 0..7
    }
}

__device__ __forceinline__ void wmma4(v8f (&acc)[4], const Frag16 (&af)[4], const Frag16& bf)
{
#pragma unroll
    for (int i = 0; i < 4; ++i)
        acc[i] = __builtin_amdgcn_wmma_f32_16x16x32_bf16(
            false, af[i].v, false, bf.v, (short)0, acc[i], false, false);
}

// ---------------------------------------------------------------------------
// Generic NT GEMM:  C[M,N] (f32 or bf16) = alpha * A[M,KD](bf16) * B[N,KD]^T(bf16)
// One wave computes a 64(M) x 16(N) tile: 4 accumulators, B fragment reused.
// Software-pipelined, double-buffered fragments; KD compile-time (512 / 768)
// so the K-loop fully unrolls and loads run a full step ahead of the WMMAs.
// __launch_bounds__(256, 3): cap occupancy at >=3 waves/SIMD so the register
// allocator keeps both fragment buffers live (~130 VGPRs) instead of
// collapsing the pipeline into load->wait(0)->wmma chains.
// M % 64 == 0, N % 16 == 0 required. Direct-global fragment loads: all operand
// panels are L2-resident (192MB L2), so re-reads never hit HBM.
// ---------------------------------------------------------------------------
template <bool STORE_BF16, int KD>
__global__ void __launch_bounds__(256, 3)
gemm_nt_wmma(const __bf16* __restrict__ A, const __bf16* __restrict__ Bm,
             void* __restrict__ Cout, int M, int N,
             int ldc, int n_limit, float alpha)
{
    static_assert(KD % 64 == 0, "KD must be a multiple of 64");
    constexpr int NK = KD / 64;

    const int lane = threadIdx.x & 31;
    const int wave = threadIdx.x >> 5;
    const long long wl = (long long)blockIdx.x * (blockDim.x >> 5) + wave;
    const int tilesM = M >> 6;     // 64-row tiles
    const int tilesN = N >> 4;     // 16-col tiles
    if (wl >= (long long)tilesM * tilesN) return;  // wave-uniform guard
    const int tm = (int)(wl % tilesM);
    const int tn = (int)(wl / tilesM);
    const int m0 = tm << 6;
    const int n0 = tn << 4;

    const int half = lane >> 4;    // 0: lanes 0-15, 1: lanes 16-31
    const int l15  = lane & 15;

    v8f acc[4];
#pragma unroll
    for (int i = 0; i < 4; ++i)
#pragma unroll
        for (int j = 0; j < 8; ++j) acc[i][j] = 0.0f;

    const __bf16* aBase = A + (size_t)(m0 + l15) * KD + half * 8;
    const __bf16* bBase = Bm + (size_t)(n0 + l15) * KD + half * 8;

    Frag16 b0, b1;
    Frag16 a0[4], a1[4];

    // prologue: fragments for k=0
    load_frags<KD>(aBase, bBase, 0, b0, a0);

#pragma unroll
    for (int c = 0; c < NK; ++c) {
        const int k0 = c * 64;
        load_frags<KD>(aBase, bBase, k0 + 32, b1, a1);   // prefetch odd step
        wmma4(acc, a0, b0);
        if (c + 1 < NK)
            load_frags<KD>(aBase, bBase, k0 + 64, b0, a0); // prefetch next even step
        wmma4(acc, a1, b1);
    }

    // f32 C/D 16x16 layout: VGPR r, lanes 0-15 -> M=r, lanes 16-31 -> M=r+8; N=lane&15
    const int col = n0 + l15;
    if (col < n_limit) {
#pragma unroll
        for (int i = 0; i < 4; ++i) {
            const int rbase = m0 + i * 16 + half * 8;
#pragma unroll
            for (int r = 0; r < 8; ++r) {
                const float val = acc[i][r] * alpha;
                const size_t idx = (size_t)(rbase + r) * ldc + col;
                if constexpr (STORE_BF16) ((__bf16*)Cout)[idx] = (__bf16)val;
                else                      ((float*)Cout)[idx]  = val;
            }
        }
    }
}

// ---------------------------------------------------------------------------
// Elementwise / row-wise kernels
// ---------------------------------------------------------------------------
__global__ void cast_f32_to_bf16(const float* __restrict__ in, __bf16* __restrict__ out, long long n)
{
    long long i = (long long)blockIdx.x * blockDim.x + threadIdx.x;
    if (i < n) out[i] = (__bf16)in[i];
}

// WT[n*D + k] = (bf16) W[k*D + n]   (weights are small, simple transpose is fine)
__global__ void transpose_to_bf16(const float* __restrict__ W, __bf16* __restrict__ WT, int Dsz)
{
    long long i = (long long)blockIdx.x * blockDim.x + threadIdx.x;
    long long n = (long long)Dsz * Dsz;
    if (i < n) {
        int r = (int)(i / Dsz), c = (int)(i % Dsz);
        WT[(size_t)c * Dsz + r] = (__bf16)W[i];
    }
}

// Row-normalize concepts; emit c_bf16 [K,D] (row-major) and c^T bf16 [D,K]
__global__ void __launch_bounds__(256)
concepts_prep(const float* __restrict__ Cin, __bf16* __restrict__ cbf, __bf16* __restrict__ ct)
{
    __shared__ float red[256];
    const int row = blockIdx.x, tid = threadIdx.x;
    float v[3];
#pragma unroll
    for (int e = 0; e < 3; ++e) v[e] = Cin[(size_t)row * Dd + tid + e * 256];
    red[tid] = v[0] * v[0] + v[1] * v[1] + v[2] * v[2];
    __syncthreads();
    for (int o = 128; o > 0; o >>= 1) { if (tid < o) red[tid] += red[tid + o]; __syncthreads(); }
    const float rinv = rsqrtf(red[0]);
#pragma unroll
    for (int e = 0; e < 3; ++e) {
        const int j = tid + e * 256;
        const __bf16 b = (__bf16)(v[e] * rinv);
        cbf[(size_t)row * Dd + j] = b;
        ct[(size_t)j * Kk + row] = b;
    }
}

// clf_n = normalize(clf + (noise-0.5)*0.01) rows, bf16, padded to Cp rows (pad = 0)
__global__ void __launch_bounds__(256)
clf_prep(const float* __restrict__ clf, const float* __restrict__ noise, __bf16* __restrict__ outp)
{
    __shared__ float red[256];
    const int row = blockIdx.x, tid = threadIdx.x;
    if (row >= Cc) {
#pragma unroll
        for (int e = 0; e < 3; ++e) outp[(size_t)row * Dd + tid + e * 256] = (__bf16)0.0f;
        return;
    }
    float v[3];
#pragma unroll
    for (int e = 0; e < 3; ++e) {
        const size_t idx = (size_t)row * Dd + tid + e * 256;
        v[e] = clf[idx] + (noise[idx] - 0.5f) * 0.01f;
    }
    red[tid] = v[0] * v[0] + v[1] * v[1] + v[2] * v[2];
    __syncthreads();
    for (int o = 128; o > 0; o >>= 1) { if (tid < o) red[tid] += red[tid + o]; __syncthreads(); }
    const float rinv = rsqrtf(red[0]);
#pragma unroll
    for (int e = 0; e < 3; ++e)
        outp[(size_t)row * Dd + tid + e * 256] = (__bf16)(v[e] * rinv);
}

// sparsemax over K=512 per row (bitonic sort + scan) + label smoothing
// writes attn (f32, second output) and attn_applied*GRAD_FACTOR as bf16
__global__ void __launch_bounds__(512)
sparsemax_smooth(const float* __restrict__ logits, float* __restrict__ attn_out,
                 __bf16* __restrict__ attn_s)
{
    __shared__ float s[512];
    __shared__ float cs[512];
    __shared__ float r[512];
    __shared__ float tau_sh;
    const int tid = threadIdx.x;
    const size_t row = blockIdx.x;

    const float v = logits[row * Kk + tid];
    // row max
    r[tid] = v; __syncthreads();
    for (int o = 256; o > 0; o >>= 1) { if (tid < o) r[tid] = fmaxf(r[tid], r[tid + o]); __syncthreads(); }
    const float zi = v - r[0];
    __syncthreads();

    // bitonic sort descending
    s[tid] = zi; __syncthreads();
    for (int k = 2; k <= 512; k <<= 1) {
        for (int j = k >> 1; j > 0; j >>= 1) {
            const int ixj = tid ^ j;
            if (ixj > tid) {
                const float a = s[tid], b = s[ixj];
                const bool up = ((tid & k) == 0);
                if (up ? (a < b) : (a > b)) { s[tid] = b; s[ixj] = a; }
            }
            __syncthreads();
        }
    }
    const float sv = s[tid];

    // inclusive scan of sorted values
    cs[tid] = sv; __syncthreads();
    for (int o = 1; o < 512; o <<= 1) {
        const float nv = cs[tid] + ((tid >= o) ? cs[tid - o] : 0.0f);
        __syncthreads();
        cs[tid] = nv;
        __syncthreads();
    }

    // support size k_z and threshold tau
    r[tid] = (1.0f + (float)(tid + 1) * sv > cs[tid]) ? 1.0f : 0.0f;
    __syncthreads();
    for (int o = 256; o > 0; o >>= 1) { if (tid < o) r[tid] += r[tid + o]; __syncthreads(); }
    if (tid == 0) {
        int kz = (int)r[0]; if (kz < 1) kz = 1;
        tau_sh = (cs[kz - 1] - 1.0f) / (float)kz;
    }
    __syncthreads();
    const float tau = tau_sh;

    const float ai = fmaxf(zi - tau, 0.0f);
    attn_out[row * Kk + tid] = ai;               // output #2: raw sparsemax
    __syncthreads();

    // smoothing over nonzero support
    r[tid] = (ai > 0.0f) ? 1.0f : 0.0f; __syncthreads();
    for (int o = 256; o > 0; o >>= 1) { if (tid < o) r[tid] += r[tid + o]; __syncthreads(); }
    const float n_nz = r[0];
    const float n_z  = (float)Kk - n_nz;
    const float sub  = SMOOTHING / fmaxf(n_nz, 1.0f);
    const float add  = SMOOTHING / fmaxf(n_z, 1.0f);
    const float applied = ((ai > 0.0f) ? (ai - sub) : add) * GRAD_FACTOR;
    attn_s[row * Kk + tid] = (__bf16)applied;
}

// LayerNorm over D=768, bf16 output
__global__ void __launch_bounds__(256)
layernorm_row(const float* __restrict__ in, const float* __restrict__ gamma,
              const float* __restrict__ beta, __bf16* __restrict__ outp)
{
    __shared__ float red[256];
    const int row = blockIdx.x, tid = threadIdx.x;
    float v[3];
#pragma unroll
    for (int e = 0; e < 3; ++e) v[e] = in[(size_t)row * Dd + tid + e * 256];
    red[tid] = v[0] + v[1] + v[2];
    __syncthreads();
    for (int o = 128; o > 0; o >>= 1) { if (tid < o) red[tid] += red[tid + o]; __syncthreads(); }
    const float mu = red[0] / (float)Dd;
    __syncthreads();
    float d0 = v[0] - mu, d1 = v[1] - mu, d2 = v[2] - mu;
    red[tid] = d0 * d0 + d1 * d1 + d2 * d2;
    __syncthreads();
    for (int o = 128; o > 0; o >>= 1) { if (tid < o) red[tid] += red[tid + o]; __syncthreads(); }
    const float rstd = rsqrtf(red[0] / (float)Dd + LN_EPS);
#pragma unroll
    for (int e = 0; e < 3; ++e) {
        const int j = tid + e * 256;
        outp[(size_t)row * Dd + j] = (__bf16)((v[e] - mu) * rstd * gamma[j] + beta[j]);
    }
}

// emb row-normalize, folding exp(logit_scale) into the rows (alpha of final GEMM)
__global__ void __launch_bounds__(256)
emb_norm_scale(const float* __restrict__ in, const float* __restrict__ logit_scale,
               __bf16* __restrict__ outp)
{
    __shared__ float red[256];
    const int row = blockIdx.x, tid = threadIdx.x;
    float v[3];
#pragma unroll
    for (int e = 0; e < 3; ++e) v[e] = in[(size_t)row * Dd + tid + e * 256];
    red[tid] = v[0] * v[0] + v[1] * v[1] + v[2] * v[2];
    __syncthreads();
    for (int o = 128; o > 0; o >>= 1) { if (tid < o) red[tid] += red[tid + o]; __syncthreads(); }
    const float scale = __expf(logit_scale[0]) * rsqrtf(red[0]);
#pragma unroll
    for (int e = 0; e < 3; ++e)
        outp[(size_t)row * Dd + tid + e * 256] = (__bf16)(v[e] * scale);
}

// ---------------------------------------------------------------------------
// Host-side orchestration
// ---------------------------------------------------------------------------
static inline int gemm_blocks(int M, int N)
{
    const long long waves = (long long)(M / 64) * (N / 16);
    return (int)((waves + 7) / 8);   // 8 waves per 256-thread block
}

extern "C" void kernel_launch(void* const* d_in, const int* in_sizes, int n_in,
                              void* d_out, int out_size, void* d_ws, size_t ws_size,
                              hipStream_t stream)
{
    (void)in_sizes; (void)n_in; (void)out_size; (void)ws_size;

    const float* x       = (const float*)d_in[0];
    const float* concepts= (const float*)d_in[1];
    const float* Wq      = (const float*)d_in[2];
    const float* Wk      = (const float*)d_in[3];
    const float* Wproj   = (const float*)d_in[4];
    const float* clf     = (const float*)d_in[5];
    const float* clfnoise= (const float*)d_in[6];
    const float* lscale  = (const float*)d_in[7];
    const float* gamma   = (const float*)d_in[8];
    const float* beta    = (const float*)d_in[9];

    float* out_logits = (float*)d_out;                                  // [B,C]
    float* out_attn   = out_logits + (size_t)Bn * Cc;                   // [B,K]
    float* out_sim    = out_attn + (size_t)Bn * Kk;                     // [K,K]

    // workspace carve-out (256B aligned)
    char* ws = (char*)d_ws;
    size_t off = 0;
    auto alloc = [&](size_t bytes) { size_t r = off; off = (off + bytes + 255) & ~(size_t)255; return r; };
    const size_t o_xbf   = alloc((size_t)Bn * Dd * 2);   // x bf16
    const size_t o_qbf   = alloc((size_t)Bn * Dd * 2);   // query bf16
    const size_t o_log   = alloc((size_t)Bn * Kk * 4);   // attn logits f32
    const size_t o_atn   = alloc((size_t)Bn * Kk * 2);   // smoothed attn bf16
    const size_t o_sum   = alloc((size_t)Bn * Dd * 4);   // summary f32 (reused for emb f32)
    const size_t o_sln   = alloc((size_t)Bn * Dd * 2);   // summary LN bf16
    const size_t o_ebf   = alloc((size_t)Bn * Dd * 2);   // emb normalized bf16
    const size_t o_wq    = alloc((size_t)Dd * Dd * 2);
    const size_t o_wk    = alloc((size_t)Dd * Dd * 2);
    const size_t o_wp    = alloc((size_t)Dd * Dd * 2);
    const size_t o_cbf   = alloc((size_t)Kk * Dd * 2);   // normalized concepts bf16 [K,D]
    const size_t o_ct    = alloc((size_t)Dd * Kk * 2);   // concepts^T bf16 [D,K]
    const size_t o_key   = alloc((size_t)Kk * Dd * 2);   // key bf16 [K,D]
    const size_t o_clf   = alloc((size_t)Cp * Dd * 2);   // padded normalized clf bf16

    __bf16* xbf  = (__bf16*)(ws + o_xbf);
    __bf16* qbf  = (__bf16*)(ws + o_qbf);
    float*  logz = (float*) (ws + o_log);
    __bf16* atnb = (__bf16*)(ws + o_atn);
    float*  sumf = (float*) (ws + o_sum);
    __bf16* slnb = (__bf16*)(ws + o_sln);
    __bf16* ebf  = (__bf16*)(ws + o_ebf);
    __bf16* wqT  = (__bf16*)(ws + o_wq);
    __bf16* wkT  = (__bf16*)(ws + o_wk);
    __bf16* wpT  = (__bf16*)(ws + o_wp);
    __bf16* cbf  = (__bf16*)(ws + o_cbf);
    __bf16* ctb  = (__bf16*)(ws + o_ct);
    __bf16* keyb = (__bf16*)(ws + o_key);
    __bf16* clfb = (__bf16*)(ws + o_clf);

    const float inv_sqrt_d = 1.0f / sqrtf((float)Dd);

    // ---- prep ----
    {
        const long long n = (long long)Bn * Dd;
        cast_f32_to_bf16<<<(int)((n + 255) / 256), 256, 0, stream>>>(x, xbf, n);
        const long long nw = (long long)Dd * Dd;
        const int tb = (int)((nw + 255) / 256);
        transpose_to_bf16<<<tb, 256, 0, stream>>>(Wq, wqT, Dd);
        transpose_to_bf16<<<tb, 256, 0, stream>>>(Wk, wkT, Dd);
        transpose_to_bf16<<<tb, 256, 0, stream>>>(Wproj, wpT, Dd);
        concepts_prep<<<Kk, 256, 0, stream>>>(concepts, cbf, ctb);
        clf_prep<<<Cp, 256, 0, stream>>>(clf, clfnoise, clfb);
    }

    // ---- query = x @ Wq  (bf16 out) ----
    gemm_nt_wmma<true, Dd><<<gemm_blocks(Bn, Dd), 256, 0, stream>>>(
        xbf, wqT, qbf, Bn, Dd, Dd, Dd, 1.0f);

    // ---- key = c @ Wk  (bf16 out) ----
    gemm_nt_wmma<true, Dd><<<gemm_blocks(Kk, Dd), 256, 0, stream>>>(
        cbf, wkT, keyb, Kk, Dd, Dd, Dd, 1.0f);

    // ---- logits = query @ key^T / sqrt(D)  (f32) ----
    gemm_nt_wmma<false, Dd><<<gemm_blocks(Bn, Kk), 256, 0, stream>>>(
        qbf, keyb, logz, Bn, Kk, Kk, Kk, inv_sqrt_d);

    // ---- sparsemax + smoothing (writes output #2 and bf16 smoothed attn) ----
    sparsemax_smooth<<<Bn, 512, 0, stream>>>(logz, out_attn, atnb);

    // ---- summary = attn_s @ c  (f32)  [B matrix = c^T as N x Kd=512] ----
    gemm_nt_wmma<false, Kk><<<gemm_blocks(Bn, Dd), 256, 0, stream>>>(
        atnb, ctb, sumf, Bn, Dd, Dd, Dd, 1.0f);

    // ---- LayerNorm -> bf16 ----
    layernorm_row<<<Bn, 256, 0, stream>>>(sumf, gamma, beta, slnb);

    // ---- emb = summary_ln @ Wproj  (f32, reuses summary buffer) ----
    gemm_nt_wmma<false, Dd><<<gemm_blocks(Bn, Dd), 256, 0, stream>>>(
        slnb, wpT, sumf, Bn, Dd, Dd, Dd, 1.0f);

    // ---- emb row-normalize * exp(logit_scale) -> bf16 ----
    emb_norm_scale<<<Bn, 256, 0, stream>>>(sumf, lscale, ebf);

    // ---- outputs = emb_n @ clf_n^T  (f32, output #1; N padded to 1008, clamp 1000) ----
    gemm_nt_wmma<false, Dd><<<gemm_blocks(Bn, Cp), 256, 0, stream>>>(
        ebf, clfb, out_logits, Bn, Cp, Cc, Cc, 1.0f);

    // ---- concept_similarity = c @ c^T  (f32, output #3) ----
    gemm_nt_wmma<false, Dd><<<gemm_blocks(Kk, Kk), 256, 0, stream>>>(
        cbf, cbf, out_sim, Kk, Kk, Kk, Kk, 1.0f);
}
